// SparseMoE_34711925686932
// MI455X (gfx1250) — matrix operand order repaired and checked by reading, not compiled, not run on hardware
//
#include <hip/hip_runtime.h>
#include <hip/hip_bf16.h>

// ---------------------------------------------------------------------------
// Sparse MoE (top-2 of 8 experts), T=2048 tokens, D=1024, F=4096.
// Weight-bandwidth-bound => bf16 WMMA (v_wmma_f32_16x16x32_bf16) with f32 accum.
// ---------------------------------------------------------------------------

#define TOKS   2048
#define DIM    1024
#define FFD    4096
#define NEXP   8
#define FCHUNK 256

typedef __attribute__((ext_vector_type(16))) __bf16 v16bf;
typedef __attribute__((ext_vector_type(8)))  __bf16 v8bf;
typedef __attribute__((ext_vector_type(8)))  float  v8f;

__device__ __forceinline__ v8f wmma_bf16(v16bf a, v16bf b, v8f c) {
    // D = A(16x32 bf16) * B(32x16 bf16) + C(16x16 f32)
    return __builtin_amdgcn_wmma_f32_16x16x32_bf16(
        /*neg_a=*/false, a, /*neg_b=*/false, b,
        /*c_mod=*/(short)0, c, /*reuse_a=*/false, /*reuse_b=*/false);
}

// A-fragment gather from an LDS row-major bf16 tile.
// ISA 16-bit A 16x32 layout: lane<16 (row M=lane) holds K {0..7,16..23};
// lane>=16 (same M) holds K {8..15,24..31}. rowbase already points at
// row*stride + kbase + half*8.
__device__ __forceinline__ v16bf load_a_frag(const __bf16* rowbase) {
    v8bf lo = *(const v8bf*)(rowbase);        // K: half*8 .. half*8+7
    v8bf hi = *(const v8bf*)(rowbase + 16);   // K: 16+half*8 .. 16+half*8+7
    return __builtin_shufflevector(lo, hi, 0,1,2,3,4,5,6,7,8,9,10,11,12,13,14,15);
}

// ---------------------------------------------------------------------------
// Workspace layout (bytes)
// ---------------------------------------------------------------------------
#define O_COUNTS 0u
#define O_TOK    (4096u)                                   // NEXP*TOKS int
#define O_GATE   (O_TOK  + NEXP*TOKS*4u)                   // NEXP*TOKS float
#define O_SLOT   (O_GATE + NEXP*TOKS*4u)                   // TOKS*2*DIM float
#define O_W1B    (O_SLOT + (size_t)TOKS*2*DIM*4u)          // E*D*F bf16
#define O_W2B    (O_W1B  + (size_t)NEXP*DIM*FFD*2u)        // E*F*D bf16

// ---------------------------------------------------------------------------
__global__ void zero_counts_kernel(int* counts) {
    if (threadIdx.x < NEXP) counts[threadIdx.x] = 0;
}

// Pack W1 (E,D,F) f32 -> bf16 B-fragment layout [e][kb=D/32][f][half][i16].
// B layout matching A: lane n<16 = column n holding K {0..7,16..23} (half=0);
// lane n+16 = column n holding K {8..15,24..31} (half=1).
__global__ __launch_bounds__(256) void prep_w1_kernel(const float* __restrict__ W1,
                                                      __bf16* __restrict__ w1b) {
    size_t idx = (size_t)blockIdx.x * blockDim.x + threadIdx.x; // E*D*F total
    int i = (int)(idx & 15);
    int h = (int)((idx >> 4) & 1);
    size_t rest  = idx >> 5;
    int f  = (int)(rest % FFD);
    size_t rest2 = rest / FFD;
    int kb = (int)(rest2 % (DIM / 32));
    int e  = (int)(rest2 / (DIM / 32));
    int d  = kb * 32 + h * 8 + (i < 8 ? i : i + 8);
    w1b[idx] = (__bf16)W1[((size_t)e * DIM + d) * FFD + f];
}

// Pack W2 (E,F,D) f32 -> bf16 B-fragment layout [e][fb=F/32][d][half][i16].
__global__ __launch_bounds__(256) void prep_w2_kernel(const float* __restrict__ W2,
                                                      __bf16* __restrict__ w2b) {
    size_t idx = (size_t)blockIdx.x * blockDim.x + threadIdx.x; // E*F*D total
    int i = (int)(idx & 15);
    int h = (int)((idx >> 4) & 1);
    size_t rest  = idx >> 5;
    int d  = (int)(rest % DIM);
    size_t rest2 = rest / DIM;
    int fb = (int)(rest2 % (FFD / 32));
    int e  = (int)(rest2 / (FFD / 32));
    int f  = fb * 32 + h * 8 + (i < 8 ? i : i + 8);
    w2b[idx] = (__bf16)W2[((size_t)e * FFD + f) * DIM + d];
}

// ---------------------------------------------------------------------------
// Router: one wave (32 lanes) per token. logits = x . Wn + bn; top-2 softmax;
// atomic append (token*2+slot, gate) to the chosen experts' lists.
// ---------------------------------------------------------------------------
__global__ __launch_bounds__(256) void router_kernel(const float* __restrict__ x,
                                                     const float* __restrict__ Wn,
                                                     const float* __restrict__ bn,
                                                     int* __restrict__ counts,
                                                     int* __restrict__ etok,
                                                     float* __restrict__ egate) {
    int wave = threadIdx.x >> 5;
    int lane = threadIdx.x & 31;
    int t = blockIdx.x * 8 + wave;
    if (t >= TOKS) return;

    float acc[NEXP];
    #pragma unroll
    for (int e = 0; e < NEXP; ++e) acc[e] = 0.f;

    const float* xr = x + (size_t)t * DIM;
    for (int d = lane; d < DIM; d += 32) {
        float xv = xr[d];
        #pragma unroll
        for (int e = 0; e < NEXP; ++e) acc[e] += xv * Wn[d * NEXP + e];
    }
    #pragma unroll
    for (int e = 0; e < NEXP; ++e) {
        #pragma unroll
        for (int off = 16; off > 0; off >>= 1)
            acc[e] += __shfl_xor(acc[e], off, 32);
    }
    if (lane == 0) {
        float v[NEXP];
        #pragma unroll
        for (int e = 0; e < NEXP; ++e) v[e] = acc[e] + bn[e];
        int i0 = 0;
        #pragma unroll
        for (int e = 1; e < NEXP; ++e) if (v[e] > v[i0]) i0 = e;
        int i1 = (i0 == 0) ? 1 : 0;
        #pragma unroll
        for (int e = 0; e < NEXP; ++e) if (e != i0 && v[e] > v[i1]) i1 = e;
        float e1 = __expf(v[i1] - v[i0]);      // stable: v[i0] is the max
        float g0 = 1.f / (1.f + e1);
        float g1 = e1 / (1.f + e1);
        int p0 = atomicAdd(&counts[i0], 1);
        etok [i0 * TOKS + p0] = t * 2 + 0;
        egate[i0 * TOKS + p0] = g0;
        int p1 = atomicAdd(&counts[i1], 1);
        etok [i1 * TOKS + p1] = t * 2 + 1;
        egate[i1 * TOKS + p1] = g1;
    }
}

// ---------------------------------------------------------------------------
// FFN: one workgroup (8 waves) per (expert, 16-token tile).
// GEMM1 (X[16x1024] @ W1 -> relu -> H chunk bf16 in LDS) fused with
// GEMM2 (H @ W2 accumulated in persistent per-wave 16x128 f32 C).
// ---------------------------------------------------------------------------
__global__ __launch_bounds__(256) void ffn_kernel(const float* __restrict__ x,
                                                  const float* __restrict__ b1,
                                                  const float* __restrict__ b2,
                                                  const __bf16* __restrict__ w1b,
                                                  const __bf16* __restrict__ w2b,
                                                  const int* __restrict__ counts,
                                                  const int* __restrict__ etok,
                                                  const float* __restrict__ egate,
                                                  float* __restrict__ slotbuf) {
    __shared__ __align__(16) __bf16 Xa[16 * DIM];      // 32 KB
    __shared__ __align__(16) __bf16 Hs[16 * FCHUNK];   // 8 KB
    __shared__ int   rowTok[16];
    __shared__ float rowGate[16];

    int e = blockIdx.x >> 7;    // 128 tiles per expert
    int m = blockIdx.x & 127;
    int cnt = counts[e];
    if (m * 16 >= cnt) return;  // uniform per block

    int tid    = threadIdx.x;
    int wave   = tid >> 5;
    int lane   = tid & 31;
    int laneLo = lane & 15;
    int hf     = lane >> 4;

    if (tid < 16) {
        int idx = m * 16 + tid;
        if (idx < cnt) {
            rowTok[tid]  = etok [e * TOKS + idx];
            rowGate[tid] = egate[e * TOKS + idx];
        } else {
            rowTok[tid]  = -1;
            rowGate[tid] = 0.f;
        }
    }
    __syncthreads();

    // Stage X tile to LDS as bf16 (row-major [16][DIM]).
    {
        int row = tid >> 4;
        int cb  = (tid & 15) * 64;
        int tk  = rowTok[row];
        if (tk >= 0) {
            const float* xr = x + (size_t)(tk >> 1) * DIM + cb;
            #pragma unroll 8
            for (int j = 0; j < 64; ++j)
                Xa[row * DIM + cb + j] = (__bf16)xr[j];
        } else {
            #pragma unroll 8
            for (int j = 0; j < 64; ++j)
                Xa[row * DIM + cb + j] = (__bf16)0.f;
        }
    }
    __syncthreads();

    const __bf16* w1e = w1b + (size_t)e * DIM * FFD;
    const __bf16* w2e = w2b + (size_t)e * FFD * DIM;
    const float*  b1e = b1 + e * FFD;
    const float*  b2e = b2 + e * DIM;

    v8f yacc[8];
    #pragma unroll
    for (int i = 0; i < 8; ++i) yacc[i] = (v8f){0.f,0.f,0.f,0.f,0.f,0.f,0.f,0.f};

    for (int fc = 0; fc < FFD; fc += FCHUNK) {
        // ---- GEMM1: H[16 x FCHUNK] = relu(X @ W1[:, fc:fc+FCHUNK] + b1) ----
        int nt0 = wave * 2;                    // this wave's two local n-tiles
        int f0  = fc + nt0 * 16;               // global F col base of tile 0
        v8f h0 = (v8f){0.f,0.f,0.f,0.f,0.f,0.f,0.f,0.f};
        v8f h1 = (v8f){0.f,0.f,0.f,0.f,0.f,0.f,0.f,0.f};
        #pragma unroll
        for (int kk = 0; kk < DIM / 32; ++kk) {
            v16bf a = load_a_frag(Xa + laneLo * DIM + kk * 32 + hf * 8);
            size_t kbrow = (size_t)kk * FFD;
            v16bf bb0 = *(const v16bf*)(w1e + (kbrow + f0 + laneLo)      * 32 + hf * 16);
            v16bf bb1 = *(const v16bf*)(w1e + (kbrow + f0 + 16 + laneLo) * 32 + hf * 16);
            h0 = wmma_bf16(a, bb0, h0);
            h1 = wmma_bf16(a, bb1, h1);
        }
        // bias + relu -> bf16 H chunk in LDS
        {
            int c0 = nt0 * 16 + laneLo;        // local col of tile 0
            float bias0 = b1e[fc + c0];
            float bias1 = b1e[fc + c0 + 16];
            #pragma unroll
            for (int r = 0; r < 8; ++r) {
                int row = r + hf * 8;          // C layout: lanes>=16 hold M=r+8
                float v0 = h0[r] + bias0; v0 = v0 > 0.f ? v0 : 0.f;
                float v1 = h1[r] + bias1; v1 = v1 > 0.f ? v1 : 0.f;
                Hs[row * FCHUNK + c0]      = (__bf16)v0;
                Hs[row * FCHUNK + c0 + 16] = (__bf16)v1;
            }
        }
        __syncthreads();

        // ---- GEMM2: yacc += H[16 x FCHUNK] @ W2[fc:fc+FCHUNK, wave's 128 cols]
        int fbBase = fc >> 5;
        int cb     = wave * 128;
        #pragma unroll
        for (int kk = 0; kk < FCHUNK / 32; ++kk) {
            v16bf a = load_a_frag(Hs + laneLo * FCHUNK + kk * 32 + hf * 8);
            size_t kbrow = (size_t)(fbBase + kk) * DIM;
            #pragma unroll
            for (int nt = 0; nt < 8; ++nt) {
                int dcol = cb + nt * 16 + laneLo;
                v16bf bb = *(const v16bf*)(w2e + (kbrow + dcol) * 32 + hf * 16);
                yacc[nt] = wmma_bf16(a, bb, yacc[nt]);
            }
        }
        // Pull next F-chunk of W2 toward the caches (global_prefetch_b8).
        if (fc + FCHUNK < FFD)
            __builtin_prefetch(w2e + ((size_t)(fbBase + FCHUNK / 32) * DIM + cb) * 32, 0, 1);
        __syncthreads();
    }

    // Write gated rows to the per-(token,slot) buffer (unique writer each).
    #pragma unroll
    for (int nt = 0; nt < 8; ++nt) {
        int col = wave * 128 + nt * 16 + laneLo;
        float bias2 = b2e[col];
        #pragma unroll
        for (int r = 0; r < 8; ++r) {
            int row = r + hf * 8;
            int tk  = rowTok[row];
            if (tk >= 0)
                slotbuf[(size_t)tk * DIM + col] = rowGate[row] * (yacc[nt][r] + bias2);
        }
    }
}

// Deterministic combine: out[t,d] = slot0 + slot1.
__global__ __launch_bounds__(256) void combine_kernel(const float* __restrict__ slotbuf,
                                                      float* __restrict__ out) {
    size_t idx = (size_t)blockIdx.x * blockDim.x + threadIdx.x; // TOKS*DIM
    size_t t = idx / DIM;
    size_t d = idx % DIM;
    out[idx] = slotbuf[(t * 2) * DIM + d] + slotbuf[(t * 2 + 1) * DIM + d];
}

// ---------------------------------------------------------------------------
extern "C" void kernel_launch(void* const* d_in, const int* in_sizes, int n_in,
                              void* d_out, int out_size, void* d_ws, size_t ws_size,
                              hipStream_t stream) {
    (void)in_sizes; (void)n_in; (void)out_size; (void)ws_size;

    const float* x  = (const float*)d_in[0];
    const float* Wn = (const float*)d_in[1];
    const float* bn = (const float*)d_in[2];
    const float* W1 = (const float*)d_in[3];
    const float* b1 = (const float*)d_in[4];
    const float* W2 = (const float*)d_in[5];
    const float* b2 = (const float*)d_in[6];
    float* out = (float*)d_out;

    char* ws = (char*)d_ws;
    int*    counts  = (int*)   (ws + O_COUNTS);
    int*    etok    = (int*)   (ws + O_TOK);
    float*  egate   = (float*) (ws + O_GATE);
    float*  slotbuf = (float*) (ws + O_SLOT);
    __bf16* w1b     = (__bf16*)(ws + O_W1B);
    __bf16* w2b     = (__bf16*)(ws + O_W2B);

    zero_counts_kernel<<<1, 32, 0, stream>>>(counts);

    const int prepBlocks = (int)(((size_t)NEXP * DIM * FFD) / 256);   // 131072
    prep_w1_kernel<<<prepBlocks, 256, 0, stream>>>(W1, w1b);
    prep_w2_kernel<<<prepBlocks, 256, 0, stream>>>(W2, w2b);

    router_kernel<<<TOKS / 8, 256, 0, stream>>>(x, Wn, bn, counts, etok, egate);

    ffn_kernel<<<NEXP * (TOKS / 16), 256, 0, stream>>>(x, b1, b2, w1b, w2b,
                                                       counts, etok, egate, slotbuf);

    combine_kernel<<<(TOKS * DIM) / 256, 256, 0, stream>>>(slotbuf, out);
}
